// mLSTM_20031727468599
// MI455X (gfx1250) — matrix operand order repaired and checked by reading, not compiled, not run on hardware
//
#include <hip/hip_runtime.h>
#include <hip/hip_bf16.h>
#include <math.h>

typedef __attribute__((ext_vector_type(16))) _Float16 v16h;
typedef __attribute__((ext_vector_type(8)))  float    v8f;
typedef __attribute__((ext_vector_type(4)))  float    v4f;

#define BDIM 4
#define SDIM 512
#define HDIM 128

// ---------------------------------------------------------------------------
// Pack W[K,N] (f32 row-major) into per-lane WMMA B-fragment order (f16).
// grid = (N/16, K/32), block = 32. Output: tile-major, 512 halfs per tile,
// lane-contiguous 16 halfs so the GEMM loads one aligned v16h per lane.
// ---------------------------------------------------------------------------
__global__ void pack_w(const float* __restrict__ W, _Float16* __restrict__ P,
                       int N) {
  const int lane = threadIdx.x;
  const int hi   = (lane >= 16) ? 1 : 0;
  const int l15  = lane & 15;
  const int tn   = blockIdx.x * 16;
  const int kt   = blockIdx.y * 32;
  v16h frag;
#pragma unroll
  for (int i = 0; i < 8; ++i) {
#pragma unroll
    for (int j = 0; j < 2; ++j) {
      int kb = ((2 * i + j) & 7) + ((2 * i + j) >> 3) * 16 + hi * 8 + kt;  // ISA 16-bit B 32x16 striping
      frag[2 * i + j] = (_Float16)W[(size_t)kb * N + tn + l15];
    }
  }
  size_t idx = (((size_t)blockIdx.y * gridDim.x + blockIdx.x) * 32 + lane) * 16;
  *(v16h*)(P + idx) = frag;
}

// A fragment: row r of A, K-window [kbase, kbase+32). Per-lane data is two
// contiguous 8-float runs -> four aligned b128 loads, then cvt to f16.
__device__ __forceinline__ v16h load_afrag(const float* __restrict__ Ar,
                                           int kbase, int hi) {
  const float* p = Ar + kbase + hi * 8;
  v4f c0 = *(const v4f*)(p);
  v4f c1 = *(const v4f*)(p + 4);
  v4f c2 = *(const v4f*)(p + 16);
  v4f c3 = *(const v4f*)(p + 20);
  v16h af;
#pragma unroll
  for (int e = 0; e < 4; ++e) {
    af[e]      = (_Float16)c0[e];
    af[4 + e]  = (_Float16)c1[e];
    af[8 + e]  = (_Float16)c2[e];
    af[12 + e] = (_Float16)c3[e];
  }
  return af;
}

// ---------------------------------------------------------------------------
// WMMA projection GEMM: Y[M,N] = A[M,K] * W[K,N] + bias[N], W pre-packed.
// One wave per 16x16 tile; K unrolled by 64 (two WMMAs in flight).
// ---------------------------------------------------------------------------
__global__ void proj_wmma(const float* __restrict__ A, int lda,
                          const _Float16* __restrict__ PW,  // packed W
                          const float* __restrict__ bias,
                          float* __restrict__ Y,            // [M,N] (ldy=N)
                          int M, int N, int K) {
  const int lane = threadIdx.x;
  const int tn   = blockIdx.x * 16;
  const int tm   = blockIdx.y * 16;
  const int hi   = (lane >= 16) ? 1 : 0;
  const int l15  = lane & 15;
  const int nT   = gridDim.x;  // N/16

  int r = tm + l15;
  if (r >= M) r = M - 1;       // clamp: invalid rows computed, never stored
  const float* Ar = A + (size_t)r * lda;

  // per-K-tile stride through packed W for this N tile
  size_t bidx = ((size_t)blockIdx.x * 32 + lane) * 16;
  const size_t bstep = (size_t)nT * 512;  // one K-tile of packed fragments

  v8f acc = {};
  for (int kt = 0; kt < K; kt += 64) {  // K is always a multiple of 64
    v16h a0 = load_afrag(Ar, kt, hi);
    v16h b0 = *(const v16h*)(PW + bidx);
    v16h a1 = load_afrag(Ar, kt + 32, hi);
    v16h b1 = *(const v16h*)(PW + bidx + bstep);
    acc = __builtin_amdgcn_wmma_f32_16x16x32_f16(false, a0, false, b0,
                                                 (short)0, acc, false, false);
    acc = __builtin_amdgcn_wmma_f32_16x16x32_f16(false, a1, false, b1,
                                                 (short)0, acc, false, false);
    bidx += 2 * bstep;
  }

  // epilogue: one bias load per lane; full tiles take the unguarded path
  const int nn = tn + l15;
  const float bv = bias[nn];
  float* Yc = Y + (size_t)(tm + hi * 8) * N + nn;
  if (tm + 16 <= M) {
#pragma unroll
    for (int rr = 0; rr < 8; ++rr) Yc[(size_t)rr * N] = acc[rr] + bv;
  } else {
#pragma unroll
    for (int rr = 0; rr < 8; ++rr)
      if (tm + rr + hi * 8 < M) Yc[(size_t)rr * N] = acc[rr] + bv;
  }
}

// ---------------------------------------------------------------------------
// Serial mLSTM scan, one workgroup per batch element. C[128][128] (64 KB f32)
// in VGPRs: 256 threads x 64 elements. 3 barriers per step; cross-lane
// reductions use wave32 shuffles.
// ---------------------------------------------------------------------------
__global__ void mlstm_scan(const float* __restrict__ ipre,  // [B*S,128]
                           const float* __restrict__ kqv,   // [B*S,384] k|q|v
                           const float* __restrict__ Wfx, const float* __restrict__ bfx,
                           const float* __restrict__ Wox, const float* __restrict__ box,
                           float* __restrict__ hseq,        // [B*S,128]
                           float* __restrict__ hlast) {     // [B,128]
  const int b   = blockIdx.x;
  const int tid = threadIdx.x;       // 256 threads = 8 waves (wave32)
  const int r   = tid >> 1;          // owned C row
  const int c0  = (tid & 1) * 64;    // owned C column base

  __shared__ float sh_h[HDIM], sh_n[HDIM], sh_f[HDIM], sh_o[HDIM];
  __shared__ float sh_i[HDIM], sh_k[HDIM], sh_q[HDIM], sh_v[HDIM];
  __shared__ float sh_part[4];
  __shared__ float sh_denom;

  float C[64];
#pragma unroll
  for (int x = 0; x < 64; ++x) C[x] = 0.0f;
  if (tid < HDIM) { sh_h[tid] = 0.0f; sh_n[tid] = 0.0f; }
  __syncthreads();

  const float inv_sqrt_h = 0.08838834764831845f;  // 1/sqrt(128)

  for (int s = 0; s < SDIM; ++s) {
    const size_t row = (size_t)b * SDIM + s;
    if (tid == 0 && s + 1 < SDIM) {   // CDNA5 global_prefetch of next step
      __builtin_prefetch(ipre + (row + 1) * HDIM, 0, 0);
      __builtin_prefetch(kqv + (row + 1) * 3 * HDIM, 0, 0);
    }
    if (tid < HDIM) {
      // f gate: recurrent matvec h_{t-1} * Wfx (column tid, coalesced)
      float a = 0.0f;
      for (int k = 0; k < HDIM; ++k) a = fmaf(sh_h[k], Wfx[k * HDIM + tid], a);
      float fv = expf(a + bfx[tid]);
      sh_f[tid] = fv;
      float iv = expf(ipre[row * HDIM + tid]);
      const float* kq = kqv + row * 3 * HDIM;
      float kv = kq[tid] * inv_sqrt_h;
      float qv = kq[HDIM + tid];
      sh_i[tid] = iv; sh_k[tid] = kv; sh_q[tid] = qv;
      sh_v[tid] = kq[2 * HDIM + tid];
      float nn = fmaf(fv, sh_n[tid], iv * kv);   // n = f*n + i*k
      sh_n[tid] = nn;
      // denom partial: wave32 shuffle reduction of n.q
      float p = nn * qv;
#pragma unroll
      for (int off = 16; off >= 1; off >>= 1) p += __shfl_xor(p, off, 32);
      if ((tid & 31) == 0) sh_part[tid >> 5] = p;
    } else {
      // o gate: recurrent matvec h_{t-1} * Wox
      int j = tid - HDIM;
      float a = 0.0f;
      for (int k = 0; k < HDIM; ++k) a = fmaf(sh_h[k], Wox[k * HDIM + j], a);
      sh_o[j] = 1.0f / (1.0f + expf(-(a + box[j])));
    }
    __syncthreads();
    if (tid == 0)
      sh_denom = fmaxf(fabsf(sh_part[0] + sh_part[1] + sh_part[2] + sh_part[3]), 1.0f);
    __syncthreads();
    // C = f (.) C + (i*v) k^T fused with rowsum of C*q (64 elems/thread)
    float fr  = sh_f[r];
    float ivr = sh_i[r] * sh_v[r];
    float acc = 0.0f;
#pragma unroll
    for (int cc = 0; cc < 64; ++cc) {
      int c = c0 + cc;
      float cv = fmaf(fr, C[cc], ivr * sh_k[c]);
      C[cc] = cv;
      acc = fmaf(cv, sh_q[c], acc);
    }
    // combine the two column-halves of each row with one lane shuffle
    acc += __shfl_xor(acc, 1, 32);
    if ((tid & 1) == 0) {
      float hv = sh_o[r] * acc / sh_denom;
      sh_h[r] = hv;
      hseq[row * HDIM + r] = hv;
      if (s == SDIM - 1) hlast[b * HDIM + r] = hv;
    }
    __syncthreads();
  }
}

// ---------------------------------------------------------------------------
extern "C" void kernel_launch(void* const* d_in, const int* in_sizes, int n_in,
                              void* d_out, int out_size, void* d_ws, size_t ws_size,
                              hipStream_t stream) {
  (void)in_sizes; (void)n_in; (void)out_size; (void)ws_size;
  const float* x     = (const float*)d_in[0];
  const float* Wix0  = (const float*)d_in[1];
  const float* bix0  = (const float*)d_in[2];
  const float* Wfx0  = (const float*)d_in[3];
  const float* bfx0  = (const float*)d_in[4];
  const float* Wox0  = (const float*)d_in[5];
  const float* box0  = (const float*)d_in[6];
  const float* Wkqv0 = (const float*)d_in[7];
  const float* bkqv0 = (const float*)d_in[8];
  const float* Wix1  = (const float*)d_in[9];
  const float* bix1  = (const float*)d_in[10];
  const float* Wfx1  = (const float*)d_in[11];
  const float* bfx1  = (const float*)d_in[12];
  const float* Wox1  = (const float*)d_in[13];
  const float* box1  = (const float*)d_in[14];
  const float* Wkqv1 = (const float*)d_in[15];
  const float* bkqv1 = (const float*)d_in[16];
  const float* Wout  = (const float*)d_in[17];
  const float* bout  = (const float*)d_in[18];

  const int M = BDIM * SDIM;  // 2048 time-batch rows
  float* ws      = (float*)d_ws;
  float* ws_i    = ws;                              // 2048*128
  float* ws_kqv  = ws_i + (size_t)M * HDIM;         // 2048*384
  float* ws_h    = ws_kqv + (size_t)M * 3 * HDIM;   // 2048*128
  float* ws_hl   = ws_h + (size_t)M * HDIM;         // 4*128
  _Float16* ws_pw = (_Float16*)(ws_hl + BDIM * HDIM);  // up to 256*384 halfs

  dim3 wv(32);
  const int mT = M / 16;

  // ---- layer 0: parallel WMMA projections of x, then serial scan ----
  pack_w<<<dim3(HDIM / 16, 256 / 32), wv, 0, stream>>>(Wix0, ws_pw, HDIM);
  proj_wmma<<<dim3(HDIM / 16, mT), wv, 0, stream>>>(x, 256, ws_pw, bix0, ws_i, M, HDIM, 256);
  pack_w<<<dim3(3 * HDIM / 16, 256 / 32), wv, 0, stream>>>(Wkqv0, ws_pw, 3 * HDIM);
  proj_wmma<<<dim3(3 * HDIM / 16, mT), wv, 0, stream>>>(x, 256, ws_pw, bkqv0, ws_kqv, M, 3 * HDIM, 256);
  mlstm_scan<<<dim3(BDIM), dim3(256), 0, stream>>>(ws_i, ws_kqv, Wfx0, bfx0, Wox0, box0, ws_h, ws_hl);

  // ---- layer 1: projections of layer-0 h sequence, then serial scan ----
  pack_w<<<dim3(HDIM / 16, HDIM / 32), wv, 0, stream>>>(Wix1, ws_pw, HDIM);
  proj_wmma<<<dim3(HDIM / 16, mT), wv, 0, stream>>>(ws_h, HDIM, ws_pw, bix1, ws_i, M, HDIM, HDIM);
  pack_w<<<dim3(3 * HDIM / 16, HDIM / 32), wv, 0, stream>>>(Wkqv1, ws_pw, 3 * HDIM);
  proj_wmma<<<dim3(3 * HDIM / 16, mT), wv, 0, stream>>>(ws_h, HDIM, ws_pw, bkqv1, ws_kqv, M, 3 * HDIM, HDIM);
  mlstm_scan<<<dim3(BDIM), dim3(256), 0, stream>>>(ws_i, ws_kqv, Wfx1, bfx1, Wox1, box1, ws_h, ws_hl);

  // ---- output projection of final hidden state: [4,128] x [128,256] ----
  pack_w<<<dim3(256 / 16, HDIM / 32), wv, 0, stream>>>(Wout, ws_pw, 256);
  proj_wmma<<<dim3(256 / 16, 1), wv, 0, stream>>>(ws_hl, HDIM, ws_pw, bout, (float*)d_out, BDIM, 256, HDIM);
}
